// SwinBlock_49194555408562
// MI455X (gfx1250) — matrix operand order repaired and checked by reading, not compiled, not run on hardware
//
#include <hip/hip_runtime.h>
#include <hip/hip_bf16.h>
#include <math.h>

typedef _Float16 half_t;
typedef __attribute__((ext_vector_type(16))) _Float16 v16h;
typedef __attribute__((ext_vector_type(8)))  _Float16 v8h;
typedef __attribute__((ext_vector_type(8)))  float    v8f;
typedef __attribute__((ext_vector_type(4)))  int      v4i;

union V16 { v16h v; v8h h2[2]; _Float16 e[16]; };

#define SHIFT 3
#define TOKENS 50176           // 16*56*56 == 1024 windows * 49
#define CDIM   256

// ---- CDNA5 async global->LDS path (ASYNCcnt), with graceful fallback -------
#if defined(__has_builtin)
#  if __has_builtin(__builtin_amdgcn_global_load_async_to_lds_b128) && \
      __has_builtin(__builtin_amdgcn_s_wait_asynccnt)
#    define HAVE_ASYNC_LDS 1
#  endif
#endif
#ifndef HAVE_ASYNC_LDS
#  define HAVE_ASYNC_LDS 0
#endif

// builtin signature (from hipcc diagnostic): arg0 = v4i __device__* (addrspace 1),
// arg1 = v4i LDS pointer (addrspace 3), then imm offset, imm cpol.
#define AS_GLOBAL(p) ((__attribute__((address_space(1))) v4i*)(p))
#define AS_LDS(p)    ((__attribute__((address_space(3))) v4i*)(p))

// ---------------------------------------------------------------------------
// Weight convert + transpose: W[K][N] fp32 -> Wt[N][K] f16
// ---------------------------------------------------------------------------
__global__ void k_convert_wt(const float* __restrict__ W, half_t* __restrict__ Wt,
                             int K, int N) {
    int idx = blockIdx.x * 256 + threadIdx.x;
    if (idx < K * N) {
        int k = idx / N, n = idx % N;
        Wt[(size_t)n * K + k] = (half_t)W[idx];
    }
}

// ---------------------------------------------------------------------------
// LayerNorm over C=256, one token per 256-thread block.
// gather=1: read from x[B,56,56,256] with shifted-window token ordering
// gather=0: identity row ordering
// ---------------------------------------------------------------------------
__global__ void k_layernorm(const float* __restrict__ src,
                            const float* __restrict__ gamma,
                            const float* __restrict__ beta,
                            half_t* __restrict__ dst, int gather) {
    __shared__ float2 red[256];
    const int t = blockIdx.x;
    const int c = threadIdx.x;
    int srcoff;
    if (gather) {
        int b  = t / 3136;
        int r2 = t % 3136;
        int w2 = r2 / 49, l = r2 % 49;
        int wy = w2 >> 3, wx = w2 & 7, py = l / 7, px = l % 7;
        int hs = (wy * 7 + py + SHIFT) % 56;
        int wc = (wx * 7 + px + SHIFT) % 56;
        srcoff = ((b * 56 + hs) * 56 + wc) * CDIM;
    } else {
        srcoff = t * CDIM;
    }
    float v = src[srcoff + c];
    red[c] = make_float2(v, v * v);
    __syncthreads();
    for (int s = 128; s > 0; s >>= 1) {
        if (c < s) { red[c].x += red[c + s].x; red[c].y += red[c + s].y; }
        __syncthreads();
    }
    float mean = red[0].x * (1.0f / 256.0f);
    float var  = red[0].y * (1.0f / 256.0f) - mean * mean;
    float y = (v - mean) * rsqrtf(var + 1.001e-5f) * gamma[c] + beta[c];
    dst[(size_t)t * CDIM + c] = (half_t)y;
}

// ---------------------------------------------------------------------------
// Generic WMMA GEMM core: C[M,N] = A[M,K] * Bt[N,K]^T, fp32 accum.
// Block: 256 threads = 8 waves; block tile 128x64; wave tile 32x32 (2x2 WMMA).
// Double-buffered LDS; next K-stage staged with async global->LDS copies
// (ASYNCcnt) overlapping the WMMA of the current stage.
// ---------------------------------------------------------------------------
template <typename Epi>
__device__ __forceinline__ void gemm_core(const half_t* __restrict__ A,
                                          const half_t* __restrict__ Bt,
                                          int K, Epi epi) {
    __shared__ half_t As[2][128 * 32];
    __shared__ half_t Bs[2][64 * 32];
    const int tid  = threadIdx.x;
    const int lane = tid & 31;
    const int wid  = tid >> 5;
    const int wm   = (wid >> 1) * 32;
    const int wn   = (wid & 1) * 32;
    const int m0   = blockIdx.y * 128;
    const int n0   = blockIdx.x * 64;
    const int hi   = lane >> 4;
    const int ln   = lane & 15;

    const int arow = tid >> 1, acol = (tid & 1) * 16;
    const int brow = tid >> 2, bcol = (tid & 3) * 8;

    v8f acc[2][2] = {};

    // stage issue: 3 x 16B per thread (A: 2, B: 1)
    auto issue = [&](int k0, int buf) {
        const half_t* ag = A  + (size_t)(m0 + arow) * K + k0 + acol;
        const half_t* bg = Bt + (size_t)(n0 + brow) * K + k0 + bcol;
#if HAVE_ASYNC_LDS
        __builtin_amdgcn_global_load_async_to_lds_b128(
            AS_GLOBAL(ag),     AS_LDS(&As[buf][arow * 32 + acol]),     0, 0);
        __builtin_amdgcn_global_load_async_to_lds_b128(
            AS_GLOBAL(ag + 8), AS_LDS(&As[buf][arow * 32 + acol + 8]), 0, 0);
        __builtin_amdgcn_global_load_async_to_lds_b128(
            AS_GLOBAL(bg),     AS_LDS(&Bs[buf][brow * 32 + bcol]),     0, 0);
#else
        *(v8h*)&As[buf][arow * 32 + acol]     = *(const v8h*)ag;
        *(v8h*)&As[buf][arow * 32 + acol + 8] = *(const v8h*)(ag + 8);
        *(v8h*)&Bs[buf][brow * 32 + bcol]     = *(const v8h*)bg;
        __builtin_prefetch(ag + 32, 0, 1);
#endif
    };

    issue(0, 0);
    int cur = 0;
    for (int k0 = 0; k0 < K; k0 += 32, cur ^= 1) {
        if (k0 + 32 < K) {
            issue(k0 + 32, cur ^ 1);       // overlap next stage with compute
#if HAVE_ASYNC_LDS
            __builtin_amdgcn_s_wait_asynccnt(3);   // current stage landed
#endif
        } else {
#if HAVE_ASYNC_LDS
            __builtin_amdgcn_s_wait_asynccnt(0);
#endif
        }
        __syncthreads();                   // all waves' stage data visible

        V16 af[2], bf[2];
#pragma unroll
        for (int mt = 0; mt < 2; ++mt) {
            const half_t* p = &As[cur][(wm + mt * 16 + ln) * 32];
            af[mt].h2[0] = *(const v8h*)(p + hi * 8);
            af[mt].h2[1] = *(const v8h*)(p + 16 + hi * 8);
        }
#pragma unroll
        for (int nt = 0; nt < 2; ++nt) {
            const half_t* p = &Bs[cur][(wn + nt * 16 + ln) * 32];
            bf[nt].h2[0] = *(const v8h*)(p + hi * 8);
            bf[nt].h2[1] = *(const v8h*)(p + 16 + hi * 8);
        }
#pragma unroll
        for (int mt = 0; mt < 2; ++mt)
#pragma unroll
            for (int nt = 0; nt < 2; ++nt)
                acc[mt][nt] = __builtin_amdgcn_wmma_f32_16x16x32_f16(
                    false, af[mt].v, false, bf[nt].v,
                    (short)0, acc[mt][nt], false, false);
        __syncthreads();                   // done reading buf before reuse
    }

#pragma unroll
    for (int mt = 0; mt < 2; ++mt)
#pragma unroll
        for (int nt = 0; nt < 2; ++nt)
#pragma unroll
            for (int r = 0; r < 8; ++r)
                epi(m0 + wm + mt * 16 + hi * 8 + r,
                    n0 + wn + nt * 16 + ln,
                    acc[mt][nt][r]);
}

// ---------------------------------------------------------------------------
// GEMM instantiations with fused epilogues
// ---------------------------------------------------------------------------
__global__ __launch_bounds__(256)
void k_gemm_qkv(const half_t* A, const half_t* Bt, const float* bias,
                half_t* q, half_t* k, half_t* v) {
    gemm_core(A, Bt, 256, [=](int row, int col, float a) {
        a += bias[col];
        int which = col >> 8, rem = col & 255;
        int head = rem >> 5, d = rem & 31;
        int win = row / 49, l = row % 49;
        size_t idx = ((size_t)(win * 8 + head) * 49 + l) * 32 + d;
        if (which == 0)      q[idx] = (half_t)(a * 0.17677669529663687f); // 1/sqrt(32)
        else if (which == 1) k[idx] = (half_t)a;
        else                 v[idx] = (half_t)a;
    });
}

__global__ __launch_bounds__(256)
void k_gemm_proj(const half_t* A, const half_t* Bt, const float* bias,
                 const float* x, float* res1) {
    gemm_core(A, Bt, 256, [=](int row, int col, float a) {
        a += bias[col];
        int b  = row / 3136;
        int r2 = row % 3136;
        int w2 = r2 / 49, l = r2 % 49;
        int wy = w2 >> 3, wx = w2 & 7, py = l / 7, px = l % 7;
        int hs = (wy * 7 + py + SHIFT) % 56;
        int wc = (wx * 7 + px + SHIFT) % 56;
        size_t dst = ((size_t)(b * 56 + hs) * 56 + wc) * CDIM + col;
        res1[dst] = x[dst] + a;      // residual + inverse roll/window scatter
    });
}

__global__ __launch_bounds__(256)
void k_gemm_fc1(const half_t* A, const half_t* Bt, const float* bias,
                half_t* out) {
    gemm_core(A, Bt, 256, [=](int row, int col, float a) {
        a += bias[col];
        float g = 0.5f * a * (1.0f + erff(a * 0.7071067811865476f)); // exact GELU
        out[(size_t)row * 1024 + col] = (half_t)g;
    });
}

__global__ __launch_bounds__(256)
void k_gemm_fc2(const half_t* A, const half_t* Bt, const float* bias,
                const float* res1, float* out) {
    gemm_core(A, Bt, 1024, [=](int row, int col, float a) {
        size_t idx = (size_t)row * CDIM + col;
        out[idx] = res1[idx] + a + bias[col];
    });
}

// ---------------------------------------------------------------------------
// Window attention: one block per (window, head). 4 waves (128 threads).
// L=49 padded to 64. QK^T (16 WMMA) -> bias+mask -> softmax -> P@V (16 WMMA).
// ---------------------------------------------------------------------------
__global__ __launch_bounds__(128)
void k_attn(const half_t* __restrict__ q, const half_t* __restrict__ k,
            const half_t* __restrict__ v, const float* __restrict__ table,
            const int* __restrict__ relidx, const float* __restrict__ mask,
            half_t* __restrict__ out) {
    __shared__ half_t Qs[64 * 32];
    __shared__ half_t Ks[64 * 32];
    __shared__ half_t Vt[32 * 64];     // transposed: [d][row]
    __shared__ float  S[64 * 64];
    __shared__ half_t Ps[64 * 64];

    const int wh   = blockIdx.x;       // win*8 + head
    const int win  = wh >> 3;
    const int head = wh & 7;
    const int w2   = win & 63;         // window index within image (mask index)
    const int tid  = threadIdx.x;
    const int lane = tid & 31;
    const int wv   = tid >> 5;
    const int base = wh * 49 * 32;

    for (int idx = tid; idx < 2048; idx += 128) {
        int row = idx >> 5, d = idx & 31;
        half_t qv = (half_t)0.f, kv = (half_t)0.f, vv = (half_t)0.f;
        if (row < 49) {
            int g = base + row * 32 + d;
            qv = q[g]; kv = k[g]; vv = v[g];
        }
        Qs[idx] = qv; Ks[idx] = kv; Vt[d * 64 + row] = vv;
    }
    __syncthreads();

    const int hi = lane >> 4, ln = lane & 15;
    const int m0 = wv * 16;

    // ---- S = Q K^T (per wave: 16 rows x 64 cols), +bias +mask, pad = -inf
    V16 aq;
    {
        const half_t* p = &Qs[(m0 + ln) * 32];
        aq.h2[0] = *(const v8h*)(p + hi * 8);
        aq.h2[1] = *(const v8h*)(p + 16 + hi * 8);
    }
#pragma unroll
    for (int nt = 0; nt < 4; ++nt) {
        V16 bk;
        const half_t* p = &Ks[(nt * 16 + ln) * 32];
        bk.h2[0] = *(const v8h*)(p + hi * 8);
        bk.h2[1] = *(const v8h*)(p + 16 + hi * 8);
        v8f acc = {};
        acc = __builtin_amdgcn_wmma_f32_16x16x32_f16(false, aq.v, false, bk.v,
                                                     (short)0, acc, false, false);
#pragma unroll
        for (int r = 0; r < 8; ++r) {
            int i = m0 + hi * 8 + r, j = nt * 16 + ln;
            float sv = acc[r];
            if (i < 49 && j < 49)
                sv += table[relidx[i * 49 + j] * 8 + head] +
                      mask[(w2 * 49 + i) * 49 + j];
            else
                sv = -1e30f;
            S[i * 64 + j] = sv;
        }
    }
    __syncthreads();

    // ---- row softmax (each wave owns rows m0..m0+15; 2 cols per lane)
    for (int r = 0; r < 16; ++r) {
        int i = m0 + r;
        float a0 = S[i * 64 + lane], a1 = S[i * 64 + 32 + lane];
        float mx = fmaxf(a0, a1);
        for (int off = 16; off > 0; off >>= 1) mx = fmaxf(mx, __shfl_xor(mx, off, 32));
        float e0 = __expf(a0 - mx), e1 = __expf(a1 - mx);
        float sm = e0 + e1;
        for (int off = 16; off > 0; off >>= 1) sm += __shfl_xor(sm, off, 32);
        float inv = 1.0f / sm;
        Ps[i * 64 + lane]      = (half_t)(e0 * inv);
        Ps[i * 64 + 32 + lane] = (half_t)(e1 * inv);
    }
    __syncthreads();

    // ---- out = P @ V : per wave 16x32, K=64 in two WMMA steps
    v8f acc2[2] = {};
#pragma unroll
    for (int ks = 0; ks < 2; ++ks) {
        V16 ap;
        const half_t* p = &Ps[(m0 + ln) * 64 + ks * 32];
        ap.h2[0] = *(const v8h*)(p + hi * 8);
        ap.h2[1] = *(const v8h*)(p + 16 + hi * 8);
#pragma unroll
        for (int nt = 0; nt < 2; ++nt) {
            V16 bv;
            const half_t* pv = &Vt[(nt * 16 + ln) * 64 + ks * 32];
            bv.h2[0] = *(const v8h*)(pv + hi * 8);
            bv.h2[1] = *(const v8h*)(pv + 16 + hi * 8);
            acc2[nt] = __builtin_amdgcn_wmma_f32_16x16x32_f16(
                false, ap.v, false, bv.v, (short)0, acc2[nt], false, false);
        }
    }
#pragma unroll
    for (int nt = 0; nt < 2; ++nt)
#pragma unroll
        for (int r = 0; r < 8; ++r) {
            int i = m0 + hi * 8 + r, d = nt * 16 + ln;
            if (i < 49)
                out[(size_t)(win * 49 + i) * CDIM + head * 32 + d] = (half_t)acc2[nt][r];
        }
}

// ---------------------------------------------------------------------------
// Launch: workspace layout (peak ~195 MiB, regions reused across phases)
// ---------------------------------------------------------------------------
extern "C" void kernel_launch(void* const* d_in, const int* in_sizes, int n_in,
                              void* d_out, int out_size, void* d_ws, size_t ws_size,
                              hipStream_t stream) {
    const float* x      = (const float*)d_in[0];
    const int*   relidx = (const int*)  d_in[3];
    const float* amask  = (const float*)d_in[4];
    const float* g1     = (const float*)d_in[5];
    const float* b1     = (const float*)d_in[6];
    const float* qkv_w  = (const float*)d_in[7];
    const float* qkv_b  = (const float*)d_in[8];
    const float* table  = (const float*)d_in[9];
    const float* proj_w = (const float*)d_in[10];
    const float* proj_b = (const float*)d_in[11];
    const float* g2     = (const float*)d_in[12];
    const float* b2     = (const float*)d_in[13];
    const float* fc1_w  = (const float*)d_in[14];
    const float* fc1_b  = (const float*)d_in[15];
    const float* fc2_w  = (const float*)d_in[16];
    const float* fc2_b  = (const float*)d_in[17];
    float* outp = (float*)d_out;

    const size_t MB = 1024 * 1024;
    char* ws = (char*)d_ws;
    float*  res1   = (float*) (ws + 0);        // 49 MiB fp32
    half_t* wqkv_t = (half_t*)(ws + 52 * MB);  // f16 transposed weights
    half_t* wproj_t= (half_t*)(ws + 53 * MB);
    half_t* wfc1_t = (half_t*)(ws + 54 * MB);
    half_t* wfc2_t = (half_t*)(ws + 55 * MB);
    half_t* xw     = (half_t*)(ws + 64 * MB);  // phase-shared: xw -> attnout -> h1
    half_t* qbuf   = (half_t*)(ws + 96 * MB);  // later reused as fc1out
    half_t* kbuf   = (half_t*)(ws + 128 * MB);
    half_t* vbuf   = (half_t*)(ws + 160 * MB);
    half_t* attnout = xw;
    half_t* h1      = xw;
    half_t* fc1out  = qbuf;

    // 0) convert + transpose weights to f16 [N][K]
    k_convert_wt<<<(256 * 768  + 255) / 256, 256, 0, stream>>>(qkv_w,  wqkv_t,  256,  768);
    k_convert_wt<<<(256 * 256  + 255) / 256, 256, 0, stream>>>(proj_w, wproj_t, 256,  256);
    k_convert_wt<<<(256 * 1024 + 255) / 256, 256, 0, stream>>>(fc1_w,  wfc1_t,  256,  1024);
    k_convert_wt<<<(1024 * 256 + 255) / 256, 256, 0, stream>>>(fc2_w,  wfc2_t,  1024, 256);

    // 1) LN1 + shifted-window gather -> f16 activations [50176, 256]
    k_layernorm<<<TOKENS, 256, 0, stream>>>(x, g1, b1, xw, 1);

    // 2) QKV GEMM [50176,256]x[256,768]
    k_gemm_qkv<<<dim3(768 / 64, TOKENS / 128), 256, 0, stream>>>(
        xw, wqkv_t, qkv_b, qbuf, kbuf, vbuf);

    // 3) per-(window, head) attention
    k_attn<<<1024 * 8, 128, 0, stream>>>(qbuf, kbuf, vbuf, table, relidx, amask, attnout);

    // 4) proj GEMM + inverse shift scatter + residual -> res1 (fp32)
    k_gemm_proj<<<dim3(256 / 64, TOKENS / 128), 256, 0, stream>>>(
        attnout, wproj_t, proj_b, x, res1);

    // 5) LN2 -> h1 f16
    k_layernorm<<<TOKENS, 256, 0, stream>>>(res1, g2, b2, h1, 0);

    // 6) fc1 + exact GELU -> fc1out f16 [50176, 1024]
    k_gemm_fc1<<<dim3(1024 / 64, TOKENS / 128), 256, 0, stream>>>(
        h1, wfc1_t, fc1_b, fc1out);

    // 7) fc2 + bias + residual -> d_out fp32
    k_gemm_fc2<<<dim3(256 / 64, TOKENS / 128), 256, 0, stream>>>(
        fc1out, wfc2_t, fc2_b, res1, outp);
}
